// SimplifiedMambaSSM_4252017623329
// MI455X (gfx1250) — matrix operand order repaired and checked by reading, not compiled, not run on hardware
//
#include <hip/hip_runtime.h>
#include <hip/hip_bf16.h>

#define D_MODEL 768
#define D_STATE 16
#define D_CONV  4
#define D_INNER 1536
#define DT_RANK 48
#define BATCH   2
#define SEQLEN  2048
#define BS      (BATCH * SEQLEN)   // 4096 rows

typedef __attribute__((ext_vector_type(16))) __bf16 v16bf;
typedef __attribute__((ext_vector_type(8)))  float  v8f;
typedef __attribute__((ext_vector_type(4)))  float  v4f;
typedef __attribute__((ext_vector_type(4)))  unsigned int v4u;
typedef __attribute__((ext_vector_type(8)))  int    v8i;
typedef __attribute__((ext_vector_type(4)))  int    v4i;

union FragBF { v4f f4[2]; v16bf v; };

__device__ __forceinline__ float sigmoidf_(float x) { return 1.0f / (1.0f + __expf(-x)); }
__device__ __forceinline__ float siluf_(float x)    { return x * sigmoidf_(x); }

#define WMMA_BF16(a, b, c) \
  __builtin_amdgcn_wmma_f32_16x16x32_bf16(false, (a), false, (b), (short)0, (c), false, false)

// ---------------------------------------------------------------------------
// Block-tiled WMMA GEMM: C[M,N] f32 = A[M,K] bf16 * B[N,K]^T bf16.
// Block tile 128(M) x 64(N), 8 waves laid out 4(M) x 2(N); each wave computes
// 32x32 = 2x2 v_wmma_f32_16x16x32_bf16 accumulators. A slab staged to LDS by
// all 256 threads; B slab (64 rows x 32 K, row stride K -> packed LDS) is a
// 2D strided tile fetched by the Tensor Data Mover (tensor_load_to_lds,
// TENSORcnt) issued by wave 0, with a cooperative-load fallback.
// Requires M%128==0, N%64==0, K%32==0.
// ---------------------------------------------------------------------------
#define BM 128
#define BN 64
#define BK 32

__global__ void gemm_bf16_wmma_tiled(const __bf16* __restrict__ A,
                                     const __bf16* __restrict__ Bw,
                                     float* __restrict__ C,
                                     int M, int N, int K,
                                     const float* __restrict__ bias) {
  __shared__ __align__(128) __bf16 sA[BM * BK];   // 8 KB, row-major stride BK
  __shared__ __align__(128) __bf16 sB[BN * BK];   // 4 KB, row-major stride BK

  const int bm   = blockIdx.x;
  const int bn   = blockIdx.y;
  const int tid  = threadIdx.x;
  const int w    = tid >> 5;
  const int lane = tid & 31;
  const int wm   = w >> 1;                // 0..3  (M direction)
  const int wn   = w & 1;                 // 0..1  (N direction)
  const int half = lane >> 4;             // 0: lanes 0-15, 1: lanes 16-31
  const int r    = lane & 15;

  const __bf16* __restrict__ Ag = A  + (size_t)(bm * BM) * K;
  const __bf16* __restrict__ Bg = Bw + (size_t)(bn * BN) * K;

  // A staging map: 2 threads per row, 32 bytes (16 bf16) each.
  const int ar = tid >> 1;
  const int ac = (tid & 1) * 16;

  v8f acc00 = {}, acc01 = {}, acc10 = {}, acc11 = {};

  for (int kc = 0; kc < K; kc += BK) {
    // ---- stage A slab (global -> VGPR -> LDS) ----
    *(v4f*)&sA[ar * BK + ac]     = *(const v4f*)(Ag + (size_t)ar * K + kc + ac);
    *(v4f*)&sA[ar * BK + ac + 8] = *(const v4f*)(Ag + (size_t)ar * K + kc + ac + 8);

    // ---- stage B slab via TDM (wave 0 issues; DMA strided-gather to LDS) ----
#if __has_builtin(__builtin_amdgcn_tensor_load_to_lds) && __has_builtin(__builtin_amdgcn_s_wait_tensorcnt)
    if (w == 0) {
      const unsigned long long ga = (unsigned long long)(uintptr_t)(Bg + kc);
      const unsigned lds_addr = (unsigned)(uintptr_t)&sB[0];   // flat low 32 bits = LDS byte addr
      v4u g0;
      g0[0] = 1u;                                   // count=1, user descriptor, no gather
      g0[1] = lds_addr;                             // D#.lds_addr
      g0[2] = (unsigned)ga;                         // D#.global_addr[31:0]
      g0[3] = (unsigned)(ga >> 32) | (2u << 30);    // D#.global_addr[56:32] | type=2 (image)
      const unsigned td0 = (unsigned)K;             // tensor_dim0 (elements along K)
      const unsigned td1 = (unsigned)N;             // tensor_dim1 (rows)
      const unsigned long long s0 = (unsigned long long)K;   // tensor_dim0_stride
      v8i g1;
      g1[0] = (int)(1u << 16);                                  // wg_mask=0, data_size=1 (2B)
      g1[1] = (int)((td0 & 0xFFFFu) << 16);                     // atomic_addr=0 | td0[15:0]
      g1[2] = (int)((td0 >> 16) | ((td1 & 0xFFFFu) << 16));     // td0[31:16] | td1[15:0]
      g1[3] = (int)((td1 >> 16) | ((unsigned)BK << 16));        // td1[31:16] | tile_dim0=32
      g1[4] = (int)BN;                                          // tile_dim1=64 | tile_dim2=0
      g1[5] = (int)(unsigned)(s0 & 0xFFFFFFFFull);              // dim0_stride[31:0]
      g1[6] = (int)(unsigned)((s0 >> 32) & 0xFFFFull);          // dim0_stride[47:32] | dim1_stride[15:0]=0
      g1[7] = 0;
      v4i gz4 = {};
      v8i gz8 = {};
      // clang-23 / amdgpu-toolchain 6-arg form:
      // (uint32x4 g0, int32x8 g1, int32x4, int32x4, int32x8, i32 cpol)
      __builtin_amdgcn_tensor_load_to_lds(g0, g1, gz4, gz4, gz8, 0);
      __builtin_amdgcn_s_wait_tensorcnt(0);
    }
#else
    // fallback: cooperative load, one v4f (8 bf16) per thread
    {
      const int i = tid;                 // BN*BK/8 == 256 chunks
      const int brow = i >> 2, bchunk = (i & 3) * 8;
      *(v4f*)&sB[brow * BK + bchunk] = *(const v4f*)(Bg + (size_t)brow * K + kc + bchunk);
    }
#endif
    __syncthreads();

    // ---- per-wave fragments from LDS (ds_load_b128), 2x2 WMMA ----
    FragBF a0, a1, b0, b1;
    const int m0r = wm * 32 + r, m1r = m0r + 16;
    a0.f4[0] = *(const v4f*)&sA[m0r * BK + half * 8];
    a0.f4[1] = *(const v4f*)&sA[m0r * BK + 16 + half * 8];
    a1.f4[0] = *(const v4f*)&sA[m1r * BK + half * 8];
    a1.f4[1] = *(const v4f*)&sA[m1r * BK + 16 + half * 8];
    const int n0r = wn * 32 + r, n1r = n0r + 16;
    b0.f4[0] = *(const v4f*)&sB[n0r * BK + half * 8];
    b0.f4[1] = *(const v4f*)&sB[n0r * BK + 16 + half * 8];
    b1.f4[0] = *(const v4f*)&sB[n1r * BK + half * 8];
    b1.f4[1] = *(const v4f*)&sB[n1r * BK + 16 + half * 8];

    acc00 = WMMA_BF16(a0.v, b0.v, acc00);
    acc01 = WMMA_BF16(a0.v, b1.v, acc01);
    acc10 = WMMA_BF16(a1.v, b0.v, acc10);
    acc11 = WMMA_BF16(a1.v, b1.v, acc11);
    __syncthreads();
  }

  // ---- epilogue: C layout col=lane%16, row=vgpr + (lane/16)*8 ----
  const int cm = bm * BM + wm * 32;
  const int cn = bn * BN + wn * 32;
  auto store_tile = [&](const v8f& a, int mt, int nt) {
    const int col = cn + nt + r;
    const float bv = bias ? bias[col] : 0.0f;
    float* cb = C + (size_t)(cm + mt + half * 8) * N + col;
#pragma unroll
    for (int i = 0; i < 8; ++i) cb[(size_t)i * N] = a[i] + bv;
  };
  store_tile(acc00, 0, 0);
  store_tile(acc01, 0, 16);
  store_tile(acc10, 16, 0);
  store_tile(acc11, 16, 16);
}

// ---------------------------------------------------------------------------
// Small-N GEMM (one wave per 16x16 tile, direct global fragments) for the
// x_proj GEMM (N=80, not divisible by 64). Layouts per CDNA5 ISA 7.12.2.
// ---------------------------------------------------------------------------
__global__ void gemm_bf16_wmma_small(const __bf16* __restrict__ A,
                                     const __bf16* __restrict__ Bw,
                                     float* __restrict__ C,
                                     int M, int N, int K,
                                     const float* __restrict__ bias) {
  const int wave   = blockIdx.x * (blockDim.x >> 5) + (threadIdx.x >> 5);
  const int lane   = threadIdx.x & 31;
  const int tilesN = N >> 4;
  const int tiles  = (M >> 4) * tilesN;
  if (wave >= tiles) return;                 // wave-uniform: EXEC stays all-ones
  const int tm   = wave / tilesN;
  const int tn   = wave - tm * tilesN;
  const int half = lane >> 4;
  const int r    = lane & 15;

  const __bf16* __restrict__ arow = A  + (size_t)(tm * 16 + r) * K;
  const __bf16* __restrict__ brow = Bw + (size_t)(tn * 16 + r) * K;

  v8f acc = {};
  for (int k = 0; k < K; k += 32) {
    FragBF fa, fb;
    fa.f4[0] = *(const v4f*)(arow + k + half * 8);
    fa.f4[1] = *(const v4f*)(arow + k + 16 + half * 8);
    fb.f4[0] = *(const v4f*)(brow + k + half * 8);
    fb.f4[1] = *(const v4f*)(brow + k + 16 + half * 8);
    if (k + 32 < K) {
      __builtin_prefetch(arow + k + 32, 0, 1);
      __builtin_prefetch(brow + k + 32, 0, 1);
    }
    acc = WMMA_BF16(fa.v, fb.v, acc);
  }

  const int   col = tn * 16 + r;
  const float bv  = bias ? bias[col] : 0.0f;
  float* cbase = C + (size_t)(tm * 16 + half * 8) * N + col;
#pragma unroll
  for (int i = 0; i < 8; ++i)
    cbase[(size_t)i * N] = acc[i] + bv;
}

// ---------------------------------------------------------------------------
// f32 -> bf16 convert with optional zero-pad.
// ---------------------------------------------------------------------------
__global__ void cvt_pad_kernel(const float* __restrict__ src, __bf16* __restrict__ dst,
                               int rows, int scols, int sstride, int dcols) {
  int idx = blockIdx.x * blockDim.x + threadIdx.x;
  if (idx >= rows * dcols) return;
  int rw = idx / dcols;
  int c  = idx - rw * dcols;
  float v = (c < scols) ? src[(size_t)rw * sstride + c] : 0.0f;
  dst[idx] = (__bf16)v;
}

// silu in-place on the x-half (cols 0..D_INNER-1) of xz (row stride 2*D_INNER)
__global__ void silu_x_kernel(float* __restrict__ xz) {
  int idx = blockIdx.x * blockDim.x + threadIdx.x;
  if (idx >= BS * D_INNER) return;
  int rw = idx / D_INNER;
  int c  = idx - rw * D_INNER;
  float* p = xz + (size_t)rw * (2 * D_INNER) + c;
  *p = siluf_(*p);
}

// depthwise causal conv (width 4) over seq + bias + silu
__global__ void conv_kernel(const float* __restrict__ xz,
                            const float* __restrict__ w,
                            const float* __restrict__ bconv,
                            float* __restrict__ Xc) {
  int idx = blockIdx.x * blockDim.x + threadIdx.x;
  if (idx >= BS * D_INNER) return;
  int rw = idx / D_INNER;
  int d  = idx - rw * D_INNER;
  int t  = rw & (SEQLEN - 1);
  float acc = bconv[d];
#pragma unroll
  for (int j = 0; j < D_CONV; ++j) {
    int tt = t - (D_CONV - 1) + j;
    if (tt >= 0)
      acc += w[d * D_CONV + j] * xz[(size_t)(rw - (D_CONV - 1) + j) * (2 * D_INNER) + d];
  }
  Xc[idx] = siluf_(acc);
}

// ---------------------------------------------------------------------------
// Sequential SSM scan. Wave32 = 2 channels x 16 states; dtA/dtBx fused.
// ---------------------------------------------------------------------------
__global__ void scan_kernel(const float* __restrict__ dtf,
                            const float* __restrict__ xdbc,
                            const float* __restrict__ Xc,
                            const float* __restrict__ A_log,
                            float* __restrict__ Y) {
  const int CPB  = 16;                               // channels per block (8 waves * 2)
  const int blk  = blockIdx.x;
  const int b    = blk / (D_INNER / CPB);
  const int cb   = (blk - b * (D_INNER / CPB)) * CPB;
  const int lane = threadIdx.x & 31;
  const int wv   = threadIdx.x >> 5;
  const int n    = lane & 15;
  const int d    = cb + wv * 2 + (lane >> 4);

  const float a = -__expf(A_log[d * D_STATE + n]);
  float h = 0.0f;

  const float* __restrict__ dtb = dtf  + (size_t)b * SEQLEN * D_INNER;
  const float* __restrict__ xb  = Xc   + (size_t)b * SEQLEN * D_INNER;
  const float* __restrict__ pb  = xdbc + (size_t)b * SEQLEN * (DT_RANK + 2 * D_STATE);
  float* __restrict__ yb        = Y    + (size_t)b * SEQLEN * D_INNER;

  for (int t = 0; t < SEQLEN; ++t) {
    const float dtv = dtb[(size_t)t * D_INNER + d];
    const float Bv  = pb[(size_t)t * 80 + DT_RANK + n];
    const float Cv  = pb[(size_t)t * 80 + DT_RANK + D_STATE + n];
    const float xv  = xb[(size_t)t * D_INNER + d];
    h = __expf(dtv * a) * h + dtv * Bv * xv;
    float yc = h * Cv;
    yc += __shfl_xor(yc, 1, 32);
    yc += __shfl_xor(yc, 2, 32);
    yc += __shfl_xor(yc, 4, 32);
    yc += __shfl_xor(yc, 8, 32);
    if (n == 0) yb[(size_t)t * D_INNER + d] = yc;
  }
}

// y * silu(z) -> bf16
__global__ void gate_kernel(const float* __restrict__ Y,
                            const float* __restrict__ xz,
                            __bf16* __restrict__ ygb) {
  int idx = blockIdx.x * blockDim.x + threadIdx.x;
  if (idx >= BS * D_INNER) return;
  int rw = idx / D_INNER;
  int c  = idx - rw * D_INNER;
  float z = xz[(size_t)rw * (2 * D_INNER) + D_INNER + c];
  ygb[idx] = (__bf16)(Y[idx] * siluf_(z));
}

// ---------------------------------------------------------------------------
extern "C" void kernel_launch(void* const* d_in, const int* in_sizes, int n_in,
                              void* d_out, int out_size, void* d_ws, size_t ws_size,
                              hipStream_t stream) {
  const float* hidden  = (const float*)d_in[0];
  const float* w_in    = (const float*)d_in[1];   // (3072, 768)
  const float* conv_w  = (const float*)d_in[2];   // (1536, 1, 4)
  const float* conv_b  = (const float*)d_in[3];
  const float* w_xp    = (const float*)d_in[4];   // (80, 1536)
  const float* w_dt    = (const float*)d_in[5];   // (1536, 48)
  const float* dt_b    = (const float*)d_in[6];
  const float* A_log   = (const float*)d_in[7];
  const float* w_out   = (const float*)d_in[8];   // (768, 1536)
  float* out = (float*)d_out;

  char*  ws  = (char*)d_ws;
  size_t off = 0;
  auto alloc = [&](size_t bytes) -> void* {
    off = (off + 255) & ~(size_t)255;
    void* p = ws + off;
    off += bytes;
    return p;
  };

  __bf16* Abf   = (__bf16*)alloc((size_t)BS * D_MODEL * 2);
  __bf16* WinB  = (__bf16*)alloc((size_t)2 * D_INNER * D_MODEL * 2);
  __bf16* WxpB  = (__bf16*)alloc((size_t)80 * D_INNER * 2);
  __bf16* WdtB  = (__bf16*)alloc((size_t)D_INNER * 64 * 2);
  __bf16* WoutB = (__bf16*)alloc((size_t)D_MODEL * D_INNER * 2);
  float*  xz    = (float*)alloc((size_t)BS * 2 * D_INNER * 4);
  float*  Xc    = (float*)alloc((size_t)BS * D_INNER * 4);
  __bf16* XcB   = (__bf16*)alloc((size_t)BS * D_INNER * 2);
  float*  xdbc  = (float*)alloc((size_t)BS * 80 * 4);
  __bf16* dtlB  = (__bf16*)alloc((size_t)BS * 64 * 2);
  float*  dtf   = (float*)alloc((size_t)BS * D_INNER * 4);
  float*  Yf    = (float*)alloc((size_t)BS * D_INNER * 4);
  __bf16* ygB   = (__bf16*)alloc((size_t)BS * D_INNER * 2);

  auto eb = [](long n) { return (int)((n + 255) / 256); };
  auto gemm_tiled = [&](const __bf16* A, const __bf16* B, float* C,
                        int M, int N, int K, const float* bias) {
    dim3 grid(M / BM, N / BN);
    gemm_bf16_wmma_tiled<<<grid, 256, 0, stream>>>(A, B, C, M, N, K, bias);
  };

  // weight / activation conversions to bf16 (dt_rank padded 48 -> 64)
  cvt_pad_kernel<<<eb((long)BS * D_MODEL), 256, 0, stream>>>(hidden, Abf, BS, D_MODEL, D_MODEL, D_MODEL);
  cvt_pad_kernel<<<eb((long)2 * D_INNER * D_MODEL), 256, 0, stream>>>(w_in, WinB, 2 * D_INNER, D_MODEL, D_MODEL, D_MODEL);
  cvt_pad_kernel<<<eb((long)80 * D_INNER), 256, 0, stream>>>(w_xp, WxpB, 80, D_INNER, D_INNER, D_INNER);
  cvt_pad_kernel<<<eb((long)D_INNER * 64), 256, 0, stream>>>(w_dt, WdtB, D_INNER, DT_RANK, DT_RANK, 64);
  cvt_pad_kernel<<<eb((long)D_MODEL * D_INNER), 256, 0, stream>>>(w_out, WoutB, D_MODEL, D_INNER, D_INNER, D_INNER);

  // 1) in_proj GEMM: xz[4096,3072] = hidden * w_in^T   (tiled, TDM-staged B)
  gemm_tiled(Abf, WinB, xz, BS, 2 * D_INNER, D_MODEL, nullptr);
  // 2) silu on x half (pre-conv)
  silu_x_kernel<<<eb((long)BS * D_INNER), 256, 0, stream>>>(xz);
  // 3) depthwise causal conv + bias + silu
  conv_kernel<<<eb((long)BS * D_INNER), 256, 0, stream>>>(xz, conv_w, conv_b, Xc);
  cvt_pad_kernel<<<eb((long)BS * D_INNER), 256, 0, stream>>>(Xc, XcB, BS, D_INNER, D_INNER, D_INNER);
  // 4) x_proj GEMM: xdbc[4096,80]  (N=80 -> small per-wave kernel)
  {
    int tiles = (BS / 16) * (80 / 16);
    gemm_bf16_wmma_small<<<(tiles + 7) / 8, 256, 0, stream>>>(XcB, WxpB, xdbc, BS, 80, D_INNER, nullptr);
  }
  // 5) dt_proj GEMM (K padded to 64), bias fused
  cvt_pad_kernel<<<eb((long)BS * 64), 256, 0, stream>>>(xdbc, dtlB, BS, DT_RANK, 80, 64);
  gemm_tiled(dtlB, WdtB, dtf, BS, D_INNER, 64, dt_b);
  // 6) fused sequential scan
  scan_kernel<<<BATCH * (D_INNER / 16), 256, 0, stream>>>(dtf, xdbc, Xc, A_log, Yf);
  // 7) gate with silu(z), convert to bf16
  gate_kernel<<<eb((long)BS * D_INNER), 256, 0, stream>>>(Yf, xz, ygB);
  // 8) out_proj GEMM straight into d_out
  gemm_tiled(ygB, WoutB, out, BS, D_MODEL, D_INNER, nullptr);
}
